// MultiHeadAttention_76055280877990
// MI455X (gfx1250) — matrix operand order repaired and checked by reading, not compiled, not run on hardware
//
#include <hip/hip_runtime.h>

// ---------------------------------------------------------------------------
// MHA forward for MI455X (gfx1250, wave32, WMMA f16->f32).
// B=4, S=2048, DM=1024, H=16, dk=64. Flash-attention (transposed tiles)
// with double-buffered GLOBAL_LOAD_ASYNC_TO_LDS staging (ASYNCcnt).
// ---------------------------------------------------------------------------

#define S_LEN   2048
#define DMODEL  1024
#define NHEAD   16
#define DKH     64
#define QB      128     // query rows per workgroup (8 waves x 16)
#define KBS     32      // keys per inner step
#define LDP     72      // LDS row pitch in f16 (64 + 8 pad), 144B rows

typedef __attribute__((ext_vector_type(16))) _Float16 v16h;
typedef __attribute__((ext_vector_type(8)))  _Float16 v8h;
typedef __attribute__((ext_vector_type(8)))  float    v8f;

static __device__ __forceinline__ v8f wmma_f16(v16h a, v16h b, v8f c) {
  // D = A(16x32) * B(32x16) + C, f32 accumulate
  return __builtin_amdgcn_wmma_f32_16x16x32_f16(false, a, false, b,
                                                (short)0, c, false, false);
}

static __device__ __forceinline__ void wait_async0() {
#if __has_builtin(__builtin_amdgcn_s_wait_asynccnt)
  __builtin_amdgcn_s_wait_asynccnt(0);
#else
  asm volatile("s_wait_asynccnt 0x0" ::: "memory");
#endif
}

// Async DMA of one 16B chunk into LDS (no VGPR round trip, ASYNCcnt-tracked).
static __device__ __forceinline__ void async_b128_to_lds(const void* gptr,
                                                         void* lptr) {
  // addrspace(3) flat pointers: low 32 bits are the LDS byte offset.
  unsigned lds_off = (unsigned)(uintptr_t)lptr;
  asm volatile("global_load_async_to_lds_b128 %0, %1, off"
               :: "v"(lds_off), "v"(gptr) : "memory");
}

// ---------------- conversion kernels ----------------
__global__ void mha_cvt_f16(const float* __restrict__ src,
                            _Float16* __restrict__ dst, int n) {
  int i = blockIdx.x * 256 + threadIdx.x;
  if (i < n) dst[i] = (_Float16)src[i];
}

// Wt[n*1024 + k] = (f16) W[k*1024 + n]   (so B-operand K is contiguous)
__global__ void mha_cvt_wT(const float* __restrict__ W,
                           _Float16* __restrict__ Wt) {
  int i = blockIdx.x * 256 + threadIdx.x;   // exactly DM*DM threads
  int n = i >> 10;
  int k = i & 1023;
  Wt[i] = (_Float16)W[k * DMODEL + n];
}

// ---------------- GEMM: Y = A(f16,[M,1024]) x W(via Wt) + bias ----------------
// Wave tile: 16(M) x 64(N). Block: 8 waves -> 128(M) x 64(N).
// grid = (M/128, 1024/64), block = 256.

__global__ void mha_gemm_qkv(const _Float16* __restrict__ A,
                             const _Float16* __restrict__ BT,   // [N,K] f16
                             const float* __restrict__ bias,
                             _Float16* __restrict__ dst) {      // [B*H,S,64]
  const int wave = threadIdx.x >> 5;
  const int lane = threadIdx.x & 31;
  const int ln   = lane & 15;
  const int g    = lane >> 4;
  const int mbase = blockIdx.x * 128 + wave * 16;
  const int nbase = blockIdx.y * 64;

  const v8f zero = {0.f,0.f,0.f,0.f,0.f,0.f,0.f,0.f};
  v8f acc[4] = {zero, zero, zero, zero};

  for (int k0 = 0; k0 < DMODEL; k0 += 32) {
    // A operand: row = mbase+ln, K(i) = k0 + 16g + i (contiguous)
    const _Float16* ap = A + (size_t)(mbase + ln) * DMODEL + k0 + g * 16;
    v8h lo = *(const v8h*)ap;
    v8h hi = *(const v8h*)(ap + 8);
    v16h a;
#pragma unroll
    for (int i = 0; i < 8; ++i) { a[i] = lo[i]; a[i + 8] = hi[i]; }
#pragma unroll
    for (int nc = 0; nc < 4; ++nc) {
      // B operand: col = nbase+nc*16+ln, K(i) = k0 + 16g + i (contiguous)
      v16h b = *(const v16h*)(BT + (size_t)(nbase + nc * 16 + ln) * DMODEL
                              + k0 + g * 16);
      acc[nc] = wmma_f16(a, b, acc[nc]);
    }
  }
  // Epilogue: D layout row M = mbase + r + 8g, col N = nbase + nc*16 + ln.
  // Scatter into [b, h, s, d] with row = b*S+s, col = h*64+d.
#pragma unroll
  for (int nc = 0; nc < 4; ++nc) {
    int col = nbase + nc * 16 + ln;
    float bv = bias[col];
    int h = col >> 6, d = col & 63;
#pragma unroll
    for (int r = 0; r < 8; ++r) {
      int row = mbase + r + 8 * g;
      int b = row >> 11, s = row & 2047;
      dst[(((size_t)b * NHEAD + h) * S_LEN + s) * DKH + d] =
          (_Float16)(acc[nc][r] + bv);
    }
  }
}

__global__ void mha_gemm_out(const _Float16* __restrict__ A,
                             const _Float16* __restrict__ BT,
                             const float* __restrict__ bias,
                             float* __restrict__ out) {        // [M,1024] f32
  const int wave = threadIdx.x >> 5;
  const int lane = threadIdx.x & 31;
  const int ln   = lane & 15;
  const int g    = lane >> 4;
  const int mbase = blockIdx.x * 128 + wave * 16;
  const int nbase = blockIdx.y * 64;

  const v8f zero = {0.f,0.f,0.f,0.f,0.f,0.f,0.f,0.f};
  v8f acc[4] = {zero, zero, zero, zero};

  for (int k0 = 0; k0 < DMODEL; k0 += 32) {
    const _Float16* ap = A + (size_t)(mbase + ln) * DMODEL + k0 + g * 16;
    v8h lo = *(const v8h*)ap;
    v8h hi = *(const v8h*)(ap + 8);
    v16h a;
#pragma unroll
    for (int i = 0; i < 8; ++i) { a[i] = lo[i]; a[i + 8] = hi[i]; }
#pragma unroll
    for (int nc = 0; nc < 4; ++nc) {
      v16h b = *(const v16h*)(BT + (size_t)(nbase + nc * 16 + ln) * DMODEL
                              + k0 + g * 16);
      acc[nc] = wmma_f16(a, b, acc[nc]);
    }
  }
#pragma unroll
  for (int nc = 0; nc < 4; ++nc) {
    int col = nbase + nc * 16 + ln;
    float bv = bias[col];
#pragma unroll
    for (int r = 0; r < 8; ++r) {
      int row = mbase + r + 8 * g;
      out[(size_t)row * DMODEL + col] = acc[nc][r] + bv;
    }
  }
}

// ---------------- Flash attention (transposed tiles) ----------------
// grid = (S/QB, B*H), block = 256 (8 waves, 16 queries each).
// S^T = K_tile x Q^T  (D: M=key, N=query -> queries live in lanes)
// O^T = V^T x P^T     (D: M=d,   N=query) -> per-lane softmax state.
// K/V tiles double-buffered in LDS via GLOBAL_LOAD_ASYNC_TO_LDS_B128.

__global__ void mha_attn(const _Float16* __restrict__ q,
                         const _Float16* __restrict__ k,
                         const _Float16* __restrict__ v,
                         _Float16* __restrict__ o16) {
  __shared__ _Float16 Ksm[2][KBS * LDP];
  __shared__ _Float16 Vsm[2][KBS * LDP];

  const int bh   = blockIdx.y;
  const int b    = bh >> 4;
  const int h    = bh & 15;
  const int tid  = threadIdx.x;
  const int wave = tid >> 5;
  const int lane = tid & 31;
  const int ln   = lane & 15;
  const int g    = lane >> 4;

  const int qbase = blockIdx.x * QB + wave * 16;
  const int myq   = qbase + ln;

  const _Float16* qp = q + (size_t)bh * S_LEN * DKH;
  const _Float16* kp = k + (size_t)bh * S_LEN * DKH;
  const _Float16* vp = v + (size_t)bh * S_LEN * DKH;

  // Per-thread staging slot: one b128 of K and one of V per step.
  const int srow = tid >> 3;
  const int sc8  = (tid & 7) * 8;

  // Q^T B-operands (2 tiles: d 0..31, 32..63), softmax scale 1/8 folded in.
  v16h Qop[2];
#pragma unroll
  for (int dt = 0; dt < 2; ++dt) {
    v16h t = *(const v16h*)(qp + (size_t)myq * DKH + dt * 32 + g * 16);
#pragma unroll
    for (int i = 0; i < 16; ++i) Qop[dt][i] = t[i] * (_Float16)0.125f;
  }

  const v8f zero = {0.f,0.f,0.f,0.f,0.f,0.f,0.f,0.f};
  v8f acc[4] = {zero, zero, zero, zero};   // O^T: 4 d-chunks of 16
  float m = -INFINITY, lsum = 0.f;

  const int kend   = blockIdx.x * QB + QB;   // causal upper bound for block
  const int nsteps = kend / KBS;

  // Prologue: async-stage step 0 into buffer 0.
  async_b128_to_lds(kp + (size_t)srow * DKH + sc8, &Ksm[0][srow * LDP + sc8]);
  async_b128_to_lds(vp + (size_t)srow * DKH + sc8, &Vsm[0][srow * LDP + sc8]);

  for (int it = 0; it < nsteps; ++it) {
    const int kb  = it * KBS;
    const int cur = it & 1;

    wait_async0();        // this wave's copies into buf[cur] are complete
    __syncthreads();      // all waves' copies done; prev compute on buf[cur^1] done

    if (kb + KBS < kend) {  // overlap next tile's DMA with this tile's WMMAs
      const int nk = kb + KBS;
      async_b128_to_lds(kp + (size_t)(nk + srow) * DKH + sc8,
                        &Ksm[cur ^ 1][srow * LDP + sc8]);
      async_b128_to_lds(vp + (size_t)(nk + srow) * DKH + sc8,
                        &Vsm[cur ^ 1][srow * LDP + sc8]);
    }

    if (kb < qbase + 16) {           // wave-uniform: EXEC stays all-ones
      const _Float16* Kc = Ksm[cur];
      const _Float16* Vc = Vsm[cur];
      // ---- S^T tiles (keys kb..kb+15 and kb+16..kb+31) ----
      float sv0[8], sv1[8];
#pragma unroll
      for (int kt = 0; kt < 2; ++kt) {
        v8f s = zero;
#pragma unroll
        for (int dt = 0; dt < 2; ++dt) {
          const _Float16* ap = Kc + (kt * 16 + ln) * LDP + dt * 32 + g * 16;
          v8h lo = *(const v8h*)ap;
          v8h hi = *(const v8h*)(ap + 8);
          v16h a;
#pragma unroll
          for (int i = 0; i < 8; ++i) { a[i] = lo[i]; a[i + 8] = hi[i]; }
          s = wmma_f16(a, Qop[dt], s);
        }
        float* sv = kt ? sv1 : sv0;
#pragma unroll
        for (int r = 0; r < 8; ++r) {
          int key = kb + kt * 16 + g * 8 + r;
          sv[r] = (key <= myq) ? s[r] : -1e30f;   // causal mask
        }
      }
      // ---- online softmax (per-query == per-lane; combine halves) ----
      float mx = -INFINITY;
#pragma unroll
      for (int r = 0; r < 8; ++r) mx = fmaxf(mx, fmaxf(sv0[r], sv1[r]));
      mx = fmaxf(mx, __shfl_xor(mx, 16, 32));
      float mnew  = fmaxf(m, mx);
      float alpha = __expf(m - mnew);
      float rs = 0.f;
      float p0[8], p1[8];
#pragma unroll
      for (int r = 0; r < 8; ++r) {
        p0[r] = __expf(sv0[r] - mnew);
        p1[r] = __expf(sv1[r] - mnew);
        rs += p0[r] + p1[r];
      }
      rs += __shfl_xor(rs, 16, 32);
      lsum = lsum * alpha + rs;
      m = mnew;
#pragma unroll
      for (int dc = 0; dc < 4; ++dc)
#pragma unroll
        for (int r = 0; r < 8; ++r) acc[dc][r] *= alpha;

      // ---- P^T into B-operand layout (half-swap shuffles) ----
      float x0[8], x1[8];
#pragma unroll
      for (int i = 0; i < 8; ++i) {
        x0[i] = __shfl_xor(p0[i], 16, 32);
        x1[i] = __shfl_xor(p1[i], 16, 32);
      }
      v16h pT;
#pragma unroll
      for (int i = 0; i < 8; ++i) {
        pT[i]     = (_Float16)(g ? x1[i] : p0[i]);   // K = 16g + i
        pT[i + 8] = (_Float16)(g ? p1[i] : x0[i]);   // K = 16g + 8 + i
      }
      // ---- O^T += V^T x P^T ----
#pragma unroll
      for (int dc = 0; dc < 4; ++dc) {
        v16h a;   // A: M = d (lane), K(i) = 16g + i
#pragma unroll
        for (int i = 0; i < 8; ++i) {
          a[i]     = Vc[(g * 16 + i) * LDP + dc * 16 + ln];
          a[i + 8] = Vc[(g * 16 + 8 + i) * LDP + dc * 16 + ln];
        }
        acc[dc] = wmma_f16(a, pT, acc[dc]);
      }
    }
  }

  // ---- epilogue: normalize and scatter to o16[b, s, h*64 + d] (f16) ----
  float inv = 1.f / lsum;
#pragma unroll
  for (int dc = 0; dc < 4; ++dc)
#pragma unroll
    for (int r = 0; r < 8; ++r) {
      int d = h * DKH + dc * 16 + 8 * g + r;
      o16[(size_t)(b * S_LEN + myq) * DMODEL + d] =
          (_Float16)(acc[dc][r] * inv);
    }
}

// ---------------------------------------------------------------------------
extern "C" void kernel_launch(void* const* d_in, const int* in_sizes, int n_in,
                              void* d_out, int out_size, void* d_ws, size_t ws_size,
                              hipStream_t stream) {
  (void)in_sizes; (void)n_in; (void)out_size; (void)ws_size;
  const float* x  = (const float*)d_in[0];
  // d_in[1] = mask (causal, computed analytically)
  const float* Wq = (const float*)d_in[2];
  const float* bq = (const float*)d_in[3];
  const float* Wk = (const float*)d_in[4];
  const float* bk = (const float*)d_in[5];
  const float* Wv = (const float*)d_in[6];
  const float* bv = (const float*)d_in[7];
  const float* Wo = (const float*)d_in[8];
  const float* bo = (const float*)d_in[9];
  float* out = (float*)d_out;

  const int M  = 4 * S_LEN;                 // 8192 rows
  const int NX = M * DMODEL;                // 8388608
  const int NW = DMODEL * DMODEL;           // 1048576

  // workspace carve (f16 elements): ~88 MB total
  _Float16* p   = (_Float16*)d_ws;
  _Float16* x16 = p;  p += NX;
  _Float16* wqt = p;  p += NW;
  _Float16* wkt = p;  p += NW;
  _Float16* wvt = p;  p += NW;
  _Float16* wot = p;  p += NW;
  _Float16* q16 = p;  p += NX;
  _Float16* k16 = p;  p += NX;
  _Float16* v16 = p;  p += NX;
  _Float16* o16 = p;

  mha_cvt_f16<<<(NX + 255) / 256, 256, 0, stream>>>(x, x16, NX);
  mha_cvt_wT <<<NW / 256, 256, 0, stream>>>(Wq, wqt);
  mha_cvt_wT <<<NW / 256, 256, 0, stream>>>(Wk, wkt);
  mha_cvt_wT <<<NW / 256, 256, 0, stream>>>(Wv, wvt);
  mha_cvt_wT <<<NW / 256, 256, 0, stream>>>(Wo, wot);

  dim3 gg(M / 128, DMODEL / 64);
  mha_gemm_qkv<<<gg, 256, 0, stream>>>(x16, wqt, bq, q16);
  mha_gemm_qkv<<<gg, 256, 0, stream>>>(x16, wkt, bk, k16);
  mha_gemm_qkv<<<gg, 256, 0, stream>>>(x16, wvt, bv, v16);

  mha_attn<<<dim3(S_LEN / QB, 4 * NHEAD), 256, 0, stream>>>(q16, k16, v16, o16);

  mha_gemm_out<<<gg, 256, 0, stream>>>(o16, wot, bo, out);
}
